// MultiScaleProteinUpscaler_67585605370567
// MI455X (gfx1250) — matrix operand order repaired and checked by reading, not compiled, not run on hardware
//
#include <hip/hip_runtime.h>
#include <hip/hip_bf16.h>

#define B_   2
#define NA_  4096
#define NR_  512
#define D_   128
#define H_   8
#define DH_  16
#define N_LOCAL 5
#define N_IPA   4
#define N_ITER  3

typedef __attribute__((ext_vector_type(16))) _Float16 v16h;
typedef __attribute__((ext_vector_type(8)))  _Float16 v8h;
typedef __attribute__((ext_vector_type(8)))  float    v8f;

__device__ __forceinline__ v8f wmma_f16(v16h a, v16h b, v8f c) {
  // D(f32,16x16) = A(f16,16x32) * B(f16,32x16) + C
  return __builtin_amdgcn_wmma_f32_16x16x32_f16(
      false, a, false, b, (short)0, c, false, false);
}
__device__ __forceinline__ v16h cat8(v8h lo, v8h hi) {
  return __builtin_shufflevector(lo, hi, 0, 1, 2, 3, 4, 5, 6, 7,
                                 8, 9, 10, 11, 12, 13, 14, 15);
}
__device__ __forceinline__ float gelu_f(float x) {
  float t = 0.7978845608028654f * (x + 0.044715f * x * x * x);
  return 0.5f * x * (1.f + tanhf(t));
}

// ---------------------------------------------------------------------------
// Deterministic atom->residue pooling (res_map is sorted per batch).
// One block per (b, r); 128 threads = one per feature d.
// ---------------------------------------------------------------------------
__global__ void pool_k(const float* __restrict__ coords, const int* __restrict__ atype,
                       const int* __restrict__ rtype, const int* __restrict__ res_map,
                       const float* __restrict__ atom_emb, const float* __restrict__ res_emb,
                       const float* __restrict__ coord_W, const float* __restrict__ coord_b,
                       float* __restrict__ feats0) {
  int br = blockIdx.x;            // b*NR + r
  int b  = br / NR_;
  int r  = br % NR_;
  int d  = threadIdx.x;
  __shared__ int seg[2];
  if (d == 0) {
    const int* rm = res_map + b * NA_;
    int lo = 0, hi = NA_;
    while (lo < hi) { int m = (lo + hi) >> 1; if (rm[m] < r) lo = m + 1; else hi = m; }
    int s0 = lo;
    lo = s0; hi = NA_;
    while (lo < hi) { int m = (lo + hi) >> 1; if (rm[m] <= r) lo = m + 1; else hi = m; }
    seg[0] = s0; seg[1] = lo;
  }
  __syncthreads();
  int s0 = seg[0], s1 = seg[1];
  float acc = 0.f;
  float w0 = coord_W[0 * D_ + d], w1 = coord_W[1 * D_ + d], w2 = coord_W[2 * D_ + d];
  float cb = coord_b[d];
  for (int a = s0; a < s1; ++a) {
    int ga = b * NA_ + a;
    acc += atom_emb[atype[ga] * D_ + d] + res_emb[rtype[ga] * D_ + d]
         + coords[ga * 3 + 0] * w0 + coords[ga * 3 + 1] * w1
         + coords[ga * 3 + 2] * w2 + cb;
  }
  feats0[(size_t)br * D_ + d] = acc / ((float)(s1 - s0) + 1e-8f);
}

// ---------------------------------------------------------------------------
// Pack fp32 weight matrices (nmat x 128 x 128, row-major [k][n]) into f16
// WMMA B-fragment-linear order so each lane's 16-half fragment is one
// contiguous 32B chunk: Wp[(((ks*8+nt)*32+lane)*16)+e] =
//   W[ks*32 + g*8 + (e&7) + (e>>3)*16][nt*16 + ln]   (lane = g*16+ln)
// ---------------------------------------------------------------------------
__global__ void pack_w_k(const float* __restrict__ W, _Float16* __restrict__ Wp, int nmat) {
  int i = blockIdx.x * blockDim.x + threadIdx.x;
  int total = nmat * D_ * D_;
  if (i >= total) return;
  int m  = i / (D_ * D_);
  int rr = i % (D_ * D_);
  int e    = rr & 15;
  int lane = (rr >> 4) & 31;
  int tile = rr >> 9;              // ks*8 + nt
  int ks = tile >> 3, nt = tile & 7;
  int ln = lane & 15, g = lane >> 4;
  int krow = ks * 32 + g * 8 + (e & 7) + (e >> 3) * 16;
  int col  = nt * 16 + ln;
  Wp[i] = (_Float16)W[(size_t)m * D_ * D_ + (size_t)krow * D_ + col];
}

// Contract pair_W (4xD) / pair_b (D) with ipa_Wb (l,D,H) -> 4x8 effective bias mats
__global__ void pairbias_k(const float* __restrict__ pair_W, const float* __restrict__ pair_b,
                           const float* __restrict__ ipa_Wb,
                           float* __restrict__ WbEff, float* __restrict__ bEff) {
  int t = blockIdx.x * blockDim.x + threadIdx.x;   // l*40 + p*8 + h, p==4 -> bias
  if (t >= N_IPA * 5 * H_) return;
  int l = t / 40, rem = t % 40, p = rem / 8, h = rem % 8;
  const float* wb = ipa_Wb + (size_t)l * D_ * H_;
  float s = 0.f;
  if (p < 4) {
    for (int d = 0; d < D_; ++d) s += pair_W[p * D_ + d] * wb[d * H_ + h];
    WbEff[(l * 4 + p) * H_ + h] = s;
  } else {
    for (int d = 0; d < D_; ++d) s += pair_b[d] * wb[d * H_ + h];
    bEff[l * H_ + h] = s;
  }
}

// ---------------------------------------------------------------------------
// Fully fused 5-layer local MLP stack. Local layers are per-residue, so one
// 256-thread block (8 waves) owns a 16-row tile; activations live in LDS the
// whole time (f16 for WMMA A-fragments via ds_load_b128, fp32 residual).
// Wave w computes column tile w each layer; barrier between layers.
// Grid: 64 blocks x 256 threads.
// ---------------------------------------------------------------------------
__global__ void local_stack_k(const float* __restrict__ F0,
                              const _Float16* __restrict__ lWp,
                              const float* __restrict__ lb,
                              float* __restrict__ Yf, _Float16* __restrict__ Yh) {
  int mt  = blockIdx.x;           // 16-row tile
  int tid = threadIdx.x;
  int wv  = tid >> 5;             // 0..7 = column tile
  int lane = tid & 31;
  int ln = lane & 15, g = lane >> 4;
  __shared__ alignas(32) _Float16 xS[16 * D_];   // 4 KB activations (f16)
  __shared__ float rS[16 * D_];                  // 8 KB residual (f32)
  for (int e = tid; e < 16 * D_; e += 256) {
    float v = F0[((size_t)mt * 16) * D_ + e];
    rS[e] = v;
    xS[e] = (_Float16)v;
  }
  __syncthreads();
  for (int l = 0; l < N_LOCAL; ++l) {
    const _Float16* Wp = lWp + (size_t)l * D_ * D_;
    v8f c = {};
#pragma unroll
    for (int ks = 0; ks < 4; ++ks) {
      int k = ks * 32;
      v8h alo = *(const v8h*)(xS + ln * D_ + k + g * 8);
      v8h ahi = *(const v8h*)(xS + ln * D_ + k + 16 + g * 8);
      v16h a = cat8(alo, ahi);
      v16h b = *(const v16h*)(Wp + (((size_t)ks * 8 + wv) * 32 + lane) * 16);
      c = wmma_f16(a, b, c);
    }
    __syncthreads();              // everyone done reading xS for this layer
    const float* bias = lb + l * D_;
#pragma unroll
    for (int v = 0; v < 8; ++v) {
      int row = v + g * 8;
      int col = wv * 16 + ln;
      float val = gelu_f(c[v] + bias[col]) + rS[row * D_ + col];
      rS[row * D_ + col] = val;
      xS[row * D_ + col] = (_Float16)val;
    }
    __syncthreads();              // new activations visible to all waves
  }
  for (int e = tid; e < 16 * D_; e += 256) {
    Yf[((size_t)mt * 16) * D_ + e] = rS[e];
    Yh[((size_t)mt * 16) * D_ + e] = xS[e];
  }
}

// ---------------------------------------------------------------------------
// WMMA GEMM (used for the IPA output projection): Y = X @ W + bo + resid,
// fp32 out + f16 row-major out. One wave per 16x16 tile, b128 fragment loads.
// Launch: 128 blocks x 128 threads.
// ---------------------------------------------------------------------------
template <bool GELU, bool RESID, bool HASB, bool WF32, int F16MODE>
__global__ void gemm_k(const _Float16* __restrict__ X, const _Float16* __restrict__ Wp,
                       const float* __restrict__ bias, const float* __restrict__ resid,
                       float* __restrict__ Yf, _Float16* __restrict__ Yh) {
  int wave = blockIdx.x * (blockDim.x >> 5) + (threadIdx.x >> 5);
  int nt = wave & 7;
  int mt = wave >> 3;
  int lane = threadIdx.x & 31;
  int ln = lane & 15, g = lane >> 4;
  v8f c = {};
  const _Float16* arow = X + ((size_t)mt * 16 + ln) * D_;
#pragma unroll
  for (int ks = 0; ks < 4; ++ks) {
    int k = ks * 32;
    v8h alo = *(const v8h*)(arow + k + g * 8);
    v8h ahi = *(const v8h*)(arow + k + 16 + g * 8);
    v16h a = cat8(alo, ahi);
    v16h b = *(const v16h*)(Wp + (((size_t)ks * 8 + nt) * 32 + lane) * 16);
    c = wmma_f16(a, b, c);
  }
#pragma unroll
  for (int v = 0; v < 8; ++v) {
    int row = mt * 16 + v + g * 8;
    int col = nt * 16 + ln;
    float val = c[v];
    if constexpr (HASB) val += bias[col];
    if constexpr (GELU) val = gelu_f(val);
    if constexpr (RESID) val += resid[(size_t)row * D_ + col];
    if constexpr (WF32) Yf[(size_t)row * D_ + col] = val;
    if constexpr (F16MODE == 1) Yh[(size_t)row * D_ + col] = (_Float16)val;
    if constexpr (F16MODE == 2) {
      int bb = row / NR_, ii = row % NR_;
      Yh[(((size_t)bb * H_ + (col >> 4)) * DH_ + (col & 15)) * NR_ + ii] = (_Float16)val;
    }
  }
}

// ---------------------------------------------------------------------------
// Fused Q/K/V projection: gridDim.z = 3 selects the weight matrix, output
// buffer and store layout (V is stored per-head transposed Vt[b][h][d][j]
// so the attention P@V B-fragment is one contiguous 16-half load).
// Launch: (128, 1, 3) x 128 threads.
// ---------------------------------------------------------------------------
__global__ void qkv_k(const _Float16* __restrict__ X,
                      const _Float16* __restrict__ Wqp, const _Float16* __restrict__ Wkp,
                      const _Float16* __restrict__ Wvp,
                      _Float16* __restrict__ q16, _Float16* __restrict__ k16,
                      _Float16* __restrict__ v16t) {
  const _Float16* Wp;
  _Float16* Y;
  int mode;
  if (blockIdx.z == 0)      { Wp = Wqp; Y = q16;  mode = 1; }
  else if (blockIdx.z == 1) { Wp = Wkp; Y = k16;  mode = 1; }
  else                      { Wp = Wvp; Y = v16t; mode = 2; }
  int wave = blockIdx.x * (blockDim.x >> 5) + (threadIdx.x >> 5);
  int nt = wave & 7;
  int mt = wave >> 3;
  int lane = threadIdx.x & 31;
  int ln = lane & 15, g = lane >> 4;
  v8f c = {};
  const _Float16* arow = X + ((size_t)mt * 16 + ln) * D_;
#pragma unroll
  for (int ks = 0; ks < 4; ++ks) {
    int k = ks * 32;
    v8h alo = *(const v8h*)(arow + k + g * 8);
    v8h ahi = *(const v8h*)(arow + k + 16 + g * 8);
    v16h a = cat8(alo, ahi);
    v16h b = *(const v16h*)(Wp + (((size_t)ks * 8 + nt) * 32 + lane) * 16);
    c = wmma_f16(a, b, c);
  }
#pragma unroll
  for (int v = 0; v < 8; ++v) {
    int row = mt * 16 + v + g * 8;
    int col = nt * 16 + ln;
    _Float16 hv = (_Float16)c[v];
    if (mode == 1) {
      Y[(size_t)row * D_ + col] = hv;
    } else {
      int bb = row / NR_, ii = row % NR_;
      Y[(((size_t)bb * H_ + (col >> 4)) * DH_ + (col & 15)) * NR_ + ii] = hv;
    }
  }
}

// ---------------------------------------------------------------------------
// Fused IPA attention for one (b, h, 16-row i-tile). Block = 128 thr (4 waves).
// Phase 1: logits = QK^T/4 + fused geometric bias (from cur_t via 4x8 Wb_eff),
//          WMMA with K padded 16->32 (padded half loads nothing).
// Phase 2: block softmax in LDS, P stored f16.
// Phase 3: O = P @ V (Vt layout), K=512 split across 4 waves, LDS reduce.
// ---------------------------------------------------------------------------
__global__ void attn_k(const _Float16* __restrict__ q16, const _Float16* __restrict__ k16,
                       const _Float16* __restrict__ v16t, const float* __restrict__ curt,
                       const float* __restrict__ res_mask,
                       const float* __restrict__ WbEff, const float* __restrict__ bEff,
                       _Float16* __restrict__ o16) {
  int it = blockIdx.x, h = blockIdx.y, b = blockIdx.z;
  int i0 = it * 16;
  int wv = threadIdx.x >> 5;
  int lane = threadIdx.x & 31;
  int ln = lane & 15, g = lane >> 4;

  __shared__ float lgS[16 * NR_];                 // 32 KB logits
  __shared__ alignas(32) _Float16 pS[16 * NR_];   // 16 KB probabilities (f16)
  __shared__ float redS[16 * 8];
  __shared__ float statS[16];
  __shared__ float partS[4 * 16 * 16];

  // Q fragment: row ln of i-tile; valid K halves = [h*16+g*8, +8), rest zero pad
  v16h afrag;
  {
    v8h z8 = {};
    v8h qlo = *(const v8h*)(q16 + ((size_t)(b * NR_ + i0 + ln)) * D_ + h * DH_ + g * 8);
    afrag = cat8(qlo, z8);
  }
  // Phase 1: each wave handles 8 of 32 j-tiles
  for (int jt = wv; jt < NR_ / 16; jt += 4) {
    int j0 = jt * 16;
    v8h zb = {};
    // K^T fragment: lane ln = column j, halves = K[j][h*16+g*8 .. +8), rest zero pad
    v16h bfrag = cat8(*(const v8h*)(k16 + ((size_t)(b * NR_ + j0 + ln)) * D_ + h * DH_ + g * 8), zb);
    v8f c = {};
    c = wmma_f16(afrag, bfrag, c);
#pragma unroll
    for (int v = 0; v < 8; ++v) {
      int irow = v + g * 8;
      int ig = i0 + irow, jg = j0 + ln;
      float rx = curt[(b * NR_ + ig) * 3 + 0] - curt[(b * NR_ + jg) * 3 + 0];
      float ry = curt[(b * NR_ + ig) * 3 + 1] - curt[(b * NR_ + jg) * 3 + 1];
      float rz = curt[(b * NR_ + ig) * 3 + 2] - curt[(b * NR_ + jg) * 3 + 2];
      float dist = sqrtf(rx * rx + ry * ry + rz * rz);
      float bias = bEff[h] + rx * WbEff[0 * H_ + h] + ry * WbEff[1 * H_ + h]
                 + rz * WbEff[2 * H_ + h] + dist * WbEff[3 * H_ + h];
      float val = c[v] * 0.25f + bias;              // 1/sqrt(DH) = 0.25
      if (res_mask[b * NR_ + jg] <= 0.f) val = -1e9f;
      lgS[irow * NR_ + jg] = val;
    }
  }
  __syncthreads();

  // Phase 2: softmax over j (16 rows x 512), 8 threads per row
  int row = threadIdx.x >> 3;
  int sgi = threadIdx.x & 7;
  float m = -3.4e38f;
  for (int j = sgi * 64; j < sgi * 64 + 64; ++j) m = fmaxf(m, lgS[row * NR_ + j]);
  redS[row * 8 + sgi] = m;
  __syncthreads();
  if (sgi == 0) {
    float mm = redS[row * 8];
    for (int s = 1; s < 8; ++s) mm = fmaxf(mm, redS[row * 8 + s]);
    statS[row] = mm;
  }
  __syncthreads();
  float rmax = statS[row];
  float sum = 0.f;
  for (int j = sgi * 64; j < sgi * 64 + 64; ++j) sum += __expf(lgS[row * NR_ + j] - rmax);
  redS[row * 8 + sgi] = sum;
  __syncthreads();
  if (sgi == 0) {
    float ss = 0.f;
    for (int s = 0; s < 8; ++s) ss += redS[row * 8 + s];
    statS[row] = ss;
  }
  __syncthreads();
  float inv = 1.f / statS[row];
  for (int j = sgi * 64; j < sgi * 64 + 64; ++j)
    pS[row * NR_ + j] = (_Float16)(__expf(lgS[row * NR_ + j] - rmax) * inv);
  __syncthreads();

  // Phase 3: O = P @ V; wave wv covers j in [wv*128, wv*128+128)
  v8f oc = {};
  int jbase = wv * 128;
  int bh = b * H_ + h;
#pragma unroll
  for (int kk = 0; kk < 128; kk += 32) {
    v16h pa = *(const v16h*)(pS + (size_t)ln * NR_ + jbase + kk + g * 16);
    v16h vb = *(const v16h*)(v16t + ((size_t)bh * DH_ + ln) * NR_ + jbase + kk + g * 16);
    oc = wmma_f16(pa, vb, oc);
  }
#pragma unroll
  for (int v = 0; v < 8; ++v)
    partS[wv * 256 + (v + g * 8) * 16 + ln] = oc[v];
  __syncthreads();
  for (int e = threadIdx.x; e < 256; e += 128) {
    int il = e >> 4, dd = e & 15;
    float acc = partS[e] + partS[256 + e] + partS[512 + e] + partS[768 + e];
    o16[((size_t)(b * NR_ + i0 + il)) * D_ + h * DH_ + dd] = (_Float16)acc;
  }
}

// ---------------------------------------------------------------------------
// Frame-update head: upd = feats @ head_W + head_b; quaternion compose.
// ---------------------------------------------------------------------------
__global__ void head_k(const float* __restrict__ feats, const float* __restrict__ headW,
                       const float* __restrict__ headb, const float* __restrict__ res_mask,
                       float* __restrict__ curR, float* __restrict__ curt) {
  int i = blockIdx.x * blockDim.x + threadIdx.x;
  if (i >= B_ * NR_) return;
  const float* f = feats + (size_t)i * D_;
  float u[6];
#pragma unroll
  for (int j = 0; j < 6; ++j) u[j] = headb[j];
  for (int k = 0; k < D_; ++k) {
    float fv = f[k];
#pragma unroll
    for (int j = 0; j < 6; ++j) u[j] += fv * headW[k * 6 + j];
  }
  float vx = u[0], vy = u[1], vz = u[2];
  float inv = rsqrtf(1.f + vx * vx + vy * vy + vz * vz);
  float w = inv, x = vx * inv, y = vy * inv, z = vz * inv;
  float Q[9] = {1 - 2 * (y * y + z * z), 2 * (x * y - w * z), 2 * (x * z + w * y),
                2 * (x * y + w * z), 1 - 2 * (x * x + z * z), 2 * (y * z - w * x),
                2 * (x * z - w * y), 2 * (y * z + w * x), 1 - 2 * (x * x + y * y)};
  float Ro[9], Rn[9];
#pragma unroll
  for (int e = 0; e < 9; ++e) Ro[e] = curR[(size_t)i * 9 + e];
#pragma unroll
  for (int r = 0; r < 3; ++r)
    for (int c = 0; c < 3; ++c) {
      float s = 0.f;
      for (int k = 0; k < 3; ++k) s += Q[r * 3 + k] * Ro[k * 3 + c];
      Rn[r * 3 + c] = s;
    }
  float tv[3] = {u[3], u[4], u[5]};
  float msk = res_mask[i];
#pragma unroll
  for (int r = 0; r < 3; ++r) {
    float s = curt[(size_t)i * 3 + r];
    for (int k = 0; k < 3; ++k) s += Rn[r * 3 + k] * tv[k];
    curt[(size_t)i * 3 + r] = s * msk;
  }
#pragma unroll
  for (int e = 0; e < 9; ++e) curR[(size_t)i * 9 + e] = Rn[e];
}

// final = cur_R^T @ (R0^T @ (x - t0)) + cur_t, per atom
__global__ void final_k(const float* __restrict__ coords, const int* __restrict__ res_map,
                        const float* __restrict__ R0all, const float* __restrict__ t0all,
                        const float* __restrict__ curR, const float* __restrict__ curt,
                        float* __restrict__ out) {
  int a = blockIdx.x * blockDim.x + threadIdx.x;
  if (a >= B_ * NA_) return;
  int b = a / NA_;
  int fr = b * NR_ + res_map[a];
  const float* R0 = R0all + (size_t)fr * 9;
  const float* t0 = t0all + (size_t)fr * 3;
  const float* Rc = curR + (size_t)fr * 9;
  const float* tc = curt + (size_t)fr * 3;
  float xm[3], loc[3];
#pragma unroll
  for (int j = 0; j < 3; ++j) xm[j] = coords[(size_t)a * 3 + j] - t0[j];
#pragma unroll
  for (int i2 = 0; i2 < 3; ++i2) {
    float s = 0.f;
    for (int j = 0; j < 3; ++j) s += R0[j * 3 + i2] * xm[j];
    loc[i2] = s;
  }
#pragma unroll
  for (int i2 = 0; i2 < 3; ++i2) {
    float s = tc[i2];
    for (int j = 0; j < 3; ++j) s += Rc[j * 3 + i2] * loc[j];
    out[(size_t)a * 3 + i2] = s;
  }
}

// ---------------------------------------------------------------------------
extern "C" void kernel_launch(void* const* d_in, const int* in_sizes, int n_in,
                              void* d_out, int out_size, void* d_ws, size_t ws_size,
                              hipStream_t stream) {
  (void)in_sizes; (void)n_in; (void)out_size; (void)ws_size;
  const float* coords    = (const float*)d_in[0];
  const int*   atype     = (const int*)d_in[1];
  const int*   rtype     = (const int*)d_in[2];
  const float* res_mask  = (const float*)d_in[4];
  const int*   res_map   = (const int*)d_in[5];
  const float* rots_bad  = (const float*)d_in[6];
  const float* trans_bad = (const float*)d_in[7];
  const float* atom_emb  = (const float*)d_in[8];
  const float* res_emb   = (const float*)d_in[9];
  const float* coord_W   = (const float*)d_in[10];
  const float* coord_b   = (const float*)d_in[11];
  const float* local_W   = (const float*)d_in[12];
  const float* local_b   = (const float*)d_in[13];
  const float* pair_W    = (const float*)d_in[14];
  const float* pair_b    = (const float*)d_in[15];
  const float* ipa_Wq    = (const float*)d_in[16];
  const float* ipa_Wk    = (const float*)d_in[17];
  const float* ipa_Wv    = (const float*)d_in[18];
  const float* ipa_Wb    = (const float*)d_in[19];
  const float* ipa_Wo    = (const float*)d_in[20];
  const float* ipa_bo    = (const float*)d_in[21];
  const float* head_W    = (const float*)d_in[22];
  const float* head_b    = (const float*)d_in[23];
  float* out = (float*)d_out;

  // ---- workspace carve-up (fp32 region, then 32B-aligned f16 region) ----
  const int FD = B_ * NR_ * D_;        // 131072
  const int DD = D_ * D_;              // 16384
  float* wsf = (float*)d_ws;
  size_t o = 0;
  float* feats0 = wsf + o; o += FD;
  float* fA     = wsf + o; o += FD;
  float* fB     = wsf + o; o += FD;
  float* curR   = wsf + o; o += B_ * NR_ * 9;
  float* curt   = wsf + o; o += B_ * NR_ * 3;
  float* WbEff  = wsf + o; o += N_IPA * 4 * H_;
  float* bEff   = wsf + o; o += N_IPA * H_;
  o = (o + 7) & ~(size_t)7;                     // 32B align for vector f16 loads
  _Float16* wsh = (_Float16*)(wsf + o);
  size_t ho = 0;
  _Float16* x16A  = wsh + ho; ho += FD;
  _Float16* x16B  = wsh + ho; ho += FD;
  _Float16* q16   = wsh + ho; ho += FD;
  _Float16* k16   = wsh + ho; ho += FD;
  _Float16* v16t  = wsh + ho; ho += FD;
  _Float16* o16   = wsh + ho; ho += FD;
  _Float16* lWp   = wsh + ho; ho += N_LOCAL * DD;
  _Float16* Wqp   = wsh + ho; ho += N_IPA * DD;
  _Float16* Wkp   = wsh + ho; ho += N_IPA * DD;
  _Float16* Wvp   = wsh + ho; ho += N_IPA * DD;
  _Float16* Wop   = wsh + ho; ho += N_IPA * DD;

  // ---- one-time per launch: pooling, weight packing, bias contraction ----
  pool_k<<<B_ * NR_, D_, 0, stream>>>(coords, atype, rtype, res_map, atom_emb,
                                      res_emb, coord_W, coord_b, feats0);
  pack_w_k<<<(N_LOCAL * DD + 255) / 256, 256, 0, stream>>>(local_W, lWp, N_LOCAL);
  pack_w_k<<<(N_IPA * DD + 255) / 256, 256, 0, stream>>>(ipa_Wq, Wqp, N_IPA);
  pack_w_k<<<(N_IPA * DD + 255) / 256, 256, 0, stream>>>(ipa_Wk, Wkp, N_IPA);
  pack_w_k<<<(N_IPA * DD + 255) / 256, 256, 0, stream>>>(ipa_Wv, Wvp, N_IPA);
  pack_w_k<<<(N_IPA * DD + 255) / 256, 256, 0, stream>>>(ipa_Wo, Wop, N_IPA);
  pairbias_k<<<1, 256, 0, stream>>>(pair_W, pair_b, ipa_Wb, WbEff, bEff);
  hipMemcpyAsync(curR, rots_bad, (size_t)B_ * NR_ * 9 * sizeof(float),
                 hipMemcpyDeviceToDevice, stream);
  hipMemcpyAsync(curt, trans_bad, (size_t)B_ * NR_ * 3 * sizeof(float),
                 hipMemcpyDeviceToDevice, stream);

  // ---- refinement iterations ----
  for (int iter = 0; iter < N_ITER; ++iter) {
    // 5-layer local stack fused in one kernel, feats = feats0 at entry
    local_stack_k<<<64, 256, 0, stream>>>(feats0, lWp, local_b, fA, x16A);
    float* bufs[2] = {fA, fB};
    _Float16* h16v[2] = {x16A, x16B};
    const float* cur = fA;
    _Float16* cur16 = x16A;
    int cb = 1;
    for (int l = 0; l < N_IPA; ++l) {
      qkv_k<<<dim3(128, 1, 3), 128, 0, stream>>>(
          cur16, Wqp + (size_t)l * DD, Wkp + (size_t)l * DD, Wvp + (size_t)l * DD,
          q16, k16, v16t);
      attn_k<<<dim3(NR_ / 16, H_, B_), 128, 0, stream>>>(
          q16, k16, v16t, curt, res_mask, WbEff + l * 32, bEff + l * 8, o16);
      gemm_k<false, true, true, true, 1><<<128, 128, 0, stream>>>(
          o16, Wop + (size_t)l * DD, ipa_bo + l * D_, cur, bufs[cb], h16v[cb]);
      cur = bufs[cb]; cur16 = h16v[cb]; cb ^= 1;
    }
    head_k<<<(B_ * NR_ + 255) / 256, 256, 0, stream>>>(cur, head_W, head_b,
                                                       res_mask, curR, curt);
  }

  // ---- outputs: cur_R | cur_t | final coords ----
  hipMemcpyAsync(out, curR, (size_t)B_ * NR_ * 9 * sizeof(float),
                 hipMemcpyDeviceToDevice, stream);
  hipMemcpyAsync(out + B_ * NR_ * 9, curt, (size_t)B_ * NR_ * 3 * sizeof(float),
                 hipMemcpyDeviceToDevice, stream);
  final_k<<<(B_ * NA_ + 255) / 256, 256, 0, stream>>>(
      coords, res_map, rots_bad, trans_bad, curR, curt,
      out + B_ * NR_ * 9 + B_ * NR_ * 3);
}
